// FLASH_90314572300662
// MI455X (gfx1250) — compile-verified, matrix operand order checked by reading, not run on hardware
//
#include <hip/hip_runtime.h>
#include <math.h>

// ---------------- problem constants (match reference) ----------------
#define B_   4
#define S_   512
#define T_   512
#define D_   512
#define DFF_ 2048
#define H_   8
#define HD_  64
#define E_   1536
#define SZ_  128
#define V_   32000

// ---------------- WMMA types ----------------
typedef __attribute__((ext_vector_type(16))) __bf16 bf16x16;
typedef __attribute__((ext_vector_type(2)))  __bf16 bf16x2;
typedef __attribute__((ext_vector_type(8)))  float  f32x8;
typedef __attribute__((ext_vector_type(2)))  float  f32x2;

union Frag { bf16x16 v; unsigned int u[8]; };

// packed fp32->bf16 (RNE) -- lowers to hardware v_cvt_pk_bf16_f32
__device__ __forceinline__ unsigned int pk2bf(float lo, float hi) {
  union { bf16x2 b; unsigned int u; } cv;
  cv.b = __builtin_convertvector((f32x2){lo, hi}, bf16x2);
  return cv.u;
}
__device__ __forceinline__ unsigned short f2bf(float f) {
  union { __bf16 b; unsigned short u; } cv;
  cv.b = (__bf16)f;
  return cv.u;
}

// gemm flags (compile-time template parameter)
#define GF_TRANSB 1
#define GF_RELU   2
#define GF_BIAS   4
#define GF_RES    8
#define GF_BF16B  16   // B is pre-converted bf16 [N,K]; staged via async copy to LDS

// ------------------------------------------------------------------
// Weight convert+transpose: fp32 [K,N] -> bf16 [N,K]
// ------------------------------------------------------------------
__global__ __launch_bounds__(256)
void convT_kernel(const float* __restrict__ Wf, unsigned short* __restrict__ Wt,
                  int K, int N)
{
  __shared__ float t[32][33];
  int kt = blockIdx.y * 32, nt = blockIdx.x * 32;
  int tx = threadIdx.x & 31, ty = threadIdx.x >> 5;   // 32 x 8
  #pragma unroll
  for (int i = 0; i < 32; i += 8)
    t[ty + i][tx] = Wf[(long)(kt + ty + i) * N + (nt + tx)];
  __syncthreads();
  #pragma unroll
  for (int i = 0; i < 32; i += 8)
    Wt[(long)(nt + ty + i) * K + (kt + tx)] = f2bf(t[tx][ty + i]);
}

// ------------------------------------------------------------------
// Generic batched GEMM: C = act(A @ B + bias [+ Res])
//   A: fp32 [M,K] (lda)
//   B: fp32 [K,N] (ldb) | fp32 [N,K] if TRANSB | bf16 [N,K] if BF16B (async-staged)
//   Tile 128x64, K-step 32; 8 wave32 waves, each = 2x2 v_wmma_f32_16x16x32_bf16
//   LDS double-buffered: next tile's global loads / async copy overlap compute.
// ------------------------------------------------------------------
template <int FLAGS>
__global__ __launch_bounds__(256)
void wmma_gemm_kernel(const float* __restrict__ A, const void* __restrict__ Bv_,
                      const float* __restrict__ bias, const float* __restrict__ Res,
                      float* __restrict__ C,
                      int M, int N, int K, int lda, int ldb, int ldc,
                      long sAb, long sBb, long sCb,
                      long sAh, long sBh, long sCh,
                      int Hn)
{
  constexpr bool BF16B  = (FLAGS & GF_BF16B)  != 0;
  constexpr bool TRANSB = (FLAGS & GF_TRANSB) != 0;

  const int z  = blockIdx.z;
  const int bz = z / Hn, hz = z % Hn;
  const long boff = (long)bz * sBb + (long)hz * sBh;
  A += (long)bz * sAb + (long)hz * sAh;
  C += (long)bz * sCb + (long)hz * sCh;
  if (FLAGS & GF_RES) Res += (long)bz * sCb + (long)hz * sCh;
  const float* Bf = (const float*)Bv_ + boff;
  const unsigned short* Bh = (const unsigned short*)Bv_ + boff;

  const int m0 = blockIdx.y * 128;
  const int n0 = blockIdx.x * 64;
  const int tid  = threadIdx.x;
  const int lane = tid & 31;
  const int wid  = tid >> 5;
  const int g = lane >> 4;        // lane group (0/1)
  const int r = lane & 15;
  const int wm = wid >> 1;        // 0..3 -> 32-row strip
  const int wn = wid & 1;         // 0..1 -> 32-col strip

  // Row stride 40 halves (80B): 16B-aligned rows (b128 LDS ops, async B128 dest)
  // and a 20-dword stride that is bank-conflict-free for the 16-lane gather.
  __shared__ __align__(16) unsigned short sA[2][128 * 40];
  __shared__ __align__(16) unsigned short sBT[2][64 * 40];

  f32x8 acc[2][2];
  #pragma unroll
  for (int i = 0; i < 2; ++i)
    #pragma unroll
    for (int j = 0; j < 2; ++j)
      acc[i][j] = (f32x8){0.f,0.f,0.f,0.f,0.f,0.f,0.f,0.f};

  // ---- staging coordinates ----
  // A: row = tid>>3 (+it*32), col4 = (tid&7)*4
  const int a_r0 = tid >> 3;
  const int a_c4 = (tid & 7) << 2;
  const float* pA0 = A + (long)(m0 + a_r0 +  0) * lda + a_c4;
  const float* pA1 = A + (long)(m0 + a_r0 + 32) * lda + a_c4;
  const float* pA2 = A + (long)(m0 + a_r0 + 64) * lda + a_c4;
  const float* pA3 = A + (long)(m0 + a_r0 + 96) * lda + a_c4;
  const int dA0 = (a_r0 +  0) * 40 + a_c4;
  const int dA1 = (a_r0 + 32) * 40 + a_c4;
  const int dA2 = (a_r0 + 64) * 40 + a_c4;
  const int dA3 = (a_r0 + 96) * 40 + a_c4;

  // B bf16 async path: row nr = tid>>2 (0..63), 16B chunk = tid&3
  const int b_nr = tid >> 2;
  const int b_ch = tid & 3;
  unsigned long long gB =
      (unsigned long long)(Bh + (long)(n0 + b_nr) * ldb + (b_ch << 3));
  const unsigned int ldsB0 =
      (unsigned int)(unsigned long long)&sBT[0][b_nr * 40] + (b_ch << 4);
  const unsigned int ldsB1 =
      (unsigned int)(unsigned long long)&sBT[1][b_nr * 40] + (b_ch << 4);

  // B fp32 transposed path
  const int bt_nr = tid >> 3;
  const int bt_c4 = (tid & 7) << 2;
  const float* pBt0 = Bf + (long)(n0 + bt_nr +  0) * ldb + bt_c4;
  const float* pBt1 = Bf + (long)(n0 + bt_nr + 32) * ldb + bt_c4;
  // B fp32 plain path
  const int bp_kr = tid >> 4;
  const int bp_c4 = (tid & 15) << 2;
  const float* pBp0 = Bf + (long)(bp_kr +  0) * ldb + n0 + bp_c4;
  const float* pBp1 = Bf + (long)(bp_kr + 16) * ldb + n0 + bp_c4;

  float4 fa0, fa1, fa2, fa3, fb0, fb1;

  auto load_tile = [&]() {
    fa0 = *(const float4*)pA0;  pA0 += 32;
    fa1 = *(const float4*)pA1;  pA1 += 32;
    fa2 = *(const float4*)pA2;  pA2 += 32;
    fa3 = *(const float4*)pA3;  pA3 += 32;
    if (!BF16B) {
      if (TRANSB) {
        fb0 = *(const float4*)pBt0;  pBt0 += 32;
        fb1 = *(const float4*)pBt1;  pBt1 += 32;
      } else {
        fb0 = *(const float4*)pBp0;  pBp0 += 32 * ldb;
        fb1 = *(const float4*)pBp1;  pBp1 += 32 * ldb;
      }
    }
  };
  auto issue_async = [&](int buf) {
    if (BF16B) {
      asm volatile("global_load_async_to_lds_b128 %0, %1, off"
                   :: "v"(buf ? ldsB1 : ldsB0), "v"(gB) : "memory");
      gB += 64;                       // advance 32 halves
    }
  };
  auto store_tile = [&](int buf) {
    unsigned short* a = sA[buf];
    *(unsigned int*)&a[dA0]     = pk2bf(fa0.x, fa0.y);
    *(unsigned int*)&a[dA0 + 2] = pk2bf(fa0.z, fa0.w);
    *(unsigned int*)&a[dA1]     = pk2bf(fa1.x, fa1.y);
    *(unsigned int*)&a[dA1 + 2] = pk2bf(fa1.z, fa1.w);
    *(unsigned int*)&a[dA2]     = pk2bf(fa2.x, fa2.y);
    *(unsigned int*)&a[dA2 + 2] = pk2bf(fa2.z, fa2.w);
    *(unsigned int*)&a[dA3]     = pk2bf(fa3.x, fa3.y);
    *(unsigned int*)&a[dA3 + 2] = pk2bf(fa3.z, fa3.w);
    if (!BF16B) {
      unsigned short* b = sBT[buf];
      if (TRANSB) {
        *(unsigned int*)&b[(bt_nr +  0) * 40 + bt_c4]     = pk2bf(fb0.x, fb0.y);
        *(unsigned int*)&b[(bt_nr +  0) * 40 + bt_c4 + 2] = pk2bf(fb0.z, fb0.w);
        *(unsigned int*)&b[(bt_nr + 32) * 40 + bt_c4]     = pk2bf(fb1.x, fb1.y);
        *(unsigned int*)&b[(bt_nr + 32) * 40 + bt_c4 + 2] = pk2bf(fb1.z, fb1.w);
      } else {
        b[(bp_c4 + 0) * 40 + bp_kr +  0] = f2bf(fb0.x);
        b[(bp_c4 + 1) * 40 + bp_kr +  0] = f2bf(fb0.y);
        b[(bp_c4 + 2) * 40 + bp_kr +  0] = f2bf(fb0.z);
        b[(bp_c4 + 3) * 40 + bp_kr +  0] = f2bf(fb0.w);
        b[(bp_c4 + 0) * 40 + bp_kr + 16] = f2bf(fb1.x);
        b[(bp_c4 + 1) * 40 + bp_kr + 16] = f2bf(fb1.y);
        b[(bp_c4 + 2) * 40 + bp_kr + 16] = f2bf(fb1.z);
        b[(bp_c4 + 3) * 40 + bp_kr + 16] = f2bf(fb1.w);
      }
    }
  };
  auto wait_async = [&]() {
    if (BF16B) asm volatile("s_wait_asynccnt 0" ::: "memory");
  };
  auto compute = [&](int buf) {
    Frag af[2], bf[2];
    #pragma unroll
    for (int mt = 0; mt < 2; ++mt) {
      const unsigned int* p =
          (const unsigned int*)&sA[buf][(wm * 32 + mt * 16 + r) * 40];
      #pragma unroll
      for (int j = 0; j < 8; ++j) {
        int kb = 2 * (j & 3) + 8 * g + 16 * (j >> 2);
        af[mt].u[j] = p[kb >> 1];
      }
    }
    #pragma unroll
    for (int nt = 0; nt < 2; ++nt) {
      const unsigned int* p =
          (const unsigned int*)&sBT[buf][(wn * 32 + nt * 16 + r) * 40];
      #pragma unroll
      for (int j = 0; j < 8; ++j) {
        int kb = 2 * (j & 3) + 8 * g + 16 * (j >> 2);
        bf[nt].u[j] = p[kb >> 1];
      }
    }
    #pragma unroll
    for (int mt = 0; mt < 2; ++mt)
      #pragma unroll
      for (int nt = 0; nt < 2; ++nt)
        acc[mt][nt] = __builtin_amdgcn_wmma_f32_16x16x32_bf16(
            false, af[mt].v, false, bf[nt].v, (short)0, acc[mt][nt], false, false);
  };

  // ---- software pipeline over K (double-buffered LDS) ----
  const int nk = K >> 5;
  issue_async(0);
  load_tile();
  store_tile(0);
  wait_async();
  __syncthreads();

  for (int ki = 0; ki < nk; ++ki) {
    const int cur = ki & 1, nxt = cur ^ 1;
    const bool hasnext = (ki + 1) < nk;
    if (hasnext) {
      issue_async(nxt);   // async B for tile ki+1 lands during compute
      load_tile();        // A (and fp32-B) global loads for tile ki+1 into regs
    }
    compute(cur);
    if (hasnext) {
      store_tile(nxt);
      wait_async();
    }
    __syncthreads();
  }

  // ---- epilogue: bias / residual / relu, fp32 store ----
  #pragma unroll
  for (int mt = 0; mt < 2; ++mt) {
    #pragma unroll
    for (int nt = 0; nt < 2; ++nt) {
      int col = n0 + wn * 32 + nt * 16 + r;
      float bv = (FLAGS & GF_BIAS) ? bias[col] : 0.0f;
      #pragma unroll
      for (int rr = 0; rr < 8; ++rr) {
        int row = m0 + wm * 32 + mt * 16 + rr + 8 * g;
        float v = acc[mt][nt][rr] + bv;
        if (FLAGS & GF_RES) v += Res[(long)row * ldc + col];
        if (FLAGS & GF_RELU) v = fmaxf(v, 0.0f);
        C[(long)row * ldc + col] = v;
      }
    }
  }
}

// host-side dispatch over the flag combinations actually used
static void launch_gemm(int flags, dim3 grid, hipStream_t st,
                        const float* A, const void* Bm, const float* bias,
                        const float* Res, float* C,
                        int M, int N, int K, int lda, int ldb, int ldc,
                        long sAb, long sBb, long sCb,
                        long sAh, long sBh, long sCh, int Hn)
{
#define LG(F) wmma_gemm_kernel<F><<<grid, 256, 0, st>>>( \
      A, Bm, bias, Res, C, M, N, K, lda, ldb, ldc, sAb, sBb, sCb, sAh, sBh, sCh, Hn)
  switch (flags) {
    case 0:                                 LG(0); break;
    case GF_TRANSB:                         LG(GF_TRANSB); break;
    case GF_BF16B | GF_BIAS:                LG(GF_BF16B | GF_BIAS); break;
    case GF_BF16B | GF_BIAS | GF_RELU:      LG(GF_BF16B | GF_BIAS | GF_RELU); break;
    case GF_BF16B | GF_BIAS | GF_RES:       LG(GF_BF16B | GF_BIAS | GF_RES); break;
    default: break;
  }
#undef LG
}

// ------------------------------------------------------------------
// LayerNorm over last dim (one block per row)
// ------------------------------------------------------------------
__global__ __launch_bounds__(256)
void layernorm_kernel(const float* __restrict__ X, const float* __restrict__ G,
                      const float* __restrict__ Bv, float* __restrict__ Y, int Dm)
{
  __shared__ float red[8];
  long row = blockIdx.x;
  const float* x = X + row * Dm;
  float* y = Y + row * Dm;
  int tid = threadIdx.x, lane = tid & 31, wid = tid >> 5;

  float s = 0.f;
  for (int i = tid; i < Dm; i += 256) s += x[i];
  for (int o = 16; o; o >>= 1) s += __shfl_down(s, o, 32);
  if (lane == 0) red[wid] = s;
  __syncthreads();
  float mean = 0.f;
  for (int i = 0; i < 8; ++i) mean += red[i];
  mean /= (float)Dm;
  __syncthreads();

  float v = 0.f;
  for (int i = tid; i < Dm; i += 256) { float d = x[i] - mean; v += d * d; }
  for (int o = 16; o; o >>= 1) v += __shfl_down(v, o, 32);
  if (lane == 0) red[wid] = v;
  __syncthreads();
  float var = 0.f;
  for (int i = 0; i < 8; ++i) var += red[i];
  var /= (float)Dm;
  float inv = rsqrtf(var + 1e-5f);
  for (int i = tid; i < Dm; i += 256) y[i] = (x[i] - mean) * inv * G[i] + Bv[i];
}

// ------------------------------------------------------------------
// Masked (+optional causal) softmax, in place; rows = (b*Hn + h)*qlen + q
// ------------------------------------------------------------------
__global__ __launch_bounds__(256)
void softmax_kernel(float* __restrict__ Sc, const int* __restrict__ ktok,
                    int klen, int qlen, int Hn, int causal, float scale)
{
  __shared__ float red[8];
  int row = blockIdx.x;
  int qpos = row % qlen;
  int b = (row / qlen) / Hn;
  float* s = Sc + (long)row * klen;
  const int* kt = ktok + (long)b * klen;
  int tid = threadIdx.x, lane = tid & 31, wid = tid >> 5;

  float mx = -3.0e38f;
  for (int j = tid; j < klen; j += 256) {
    float v = s[j];
    if (kt[j] == -1 || (causal && j > qpos)) v = -1.0e9f;
    v *= scale;
    mx = fmaxf(mx, v);
  }
  for (int o = 16; o; o >>= 1) mx = fmaxf(mx, __shfl_down(mx, o, 32));
  if (lane == 0) red[wid] = mx;
  __syncthreads();
  float rmax = red[0];
  for (int i = 1; i < 8; ++i) rmax = fmaxf(rmax, red[i]);
  __syncthreads();

  float sum = 0.f;
  for (int j = tid; j < klen; j += 256) {
    float v = s[j];
    if (kt[j] == -1 || (causal && j > qpos)) v = -1.0e9f;
    v *= scale;
    float e = __expf(v - rmax);
    s[j] = e;
    sum += e;
  }
  for (int o = 16; o; o >>= 1) sum += __shfl_down(sum, o, 32);
  if (lane == 0) red[wid] = sum;
  __syncthreads();
  float tot = 0.f;
  for (int i = 0; i < 8; ++i) tot += red[i];
  float inv = 1.0f / tot;
  for (int j = tid; j < klen; j += 256) s[j] *= inv;
}

// ------------------------------------------------------------------
// Elementwise helpers
// ------------------------------------------------------------------
__global__ void embed_kernel(const int* __restrict__ tok, const float* __restrict__ emb,
                             const float* __restrict__ pe, float* __restrict__ out,
                             int Seq, int Dm, float scale, long n)
{
  long i = (long)blockIdx.x * blockDim.x + threadIdx.x;
  if (i >= n) return;
  int d = (int)(i % Dm);
  long bs = i / Dm;
  int sp = (int)(bs % Seq);
  int t = tok[bs];
  out[i] = emb[(long)t * Dm + d] * scale + pe[(long)sp * Dm + d];
}

__global__ void qkgate_kernel(const float* __restrict__ z,
                              const float* __restrict__ gq, const float* __restrict__ bq,
                              const float* __restrict__ gk, const float* __restrict__ bk,
                              float* __restrict__ q, float* __restrict__ k, long n)
{
  long i = (long)blockIdx.x * blockDim.x + threadIdx.x;
  if (i >= n) return;
  long p = i % ((long)S_ * SZ_);     // (s, c) parameter index
  float zv = z[i];
  q[i] = zv * gq[p] + bq[p];
  k[i] = zv * gk[p] + bk[p];
}

__global__ void relusq_mask_kernel(float* __restrict__ a, const int* __restrict__ stok, long n)
{
  long i = (long)blockIdx.x * blockDim.x + threadIdx.x;
  if (i >= n) return;
  int j = (int)(i % S_);
  int b = (int)(i / ((long)S_ * S_));
  if (stok[(long)b * S_ + j] == -1) { a[i] = 0.f; return; }
  float v = fmaxf(a[i], 0.f);
  a[i] = v * v;
}

__global__ void mul_kernel(const float* __restrict__ a, const float* __restrict__ b,
                           float* __restrict__ c, long n)
{
  long i = (long)blockIdx.x * blockDim.x + threadIdx.x;
  if (i >= n) return;
  c[i] = a[i] * b[i];
}

// ------------------------------------------------------------------
// Host orchestration
// ------------------------------------------------------------------
extern "C" void kernel_launch(void* const* d_in, const int* in_sizes, int n_in,
                              void* d_out, int out_size, void* d_ws, size_t ws_size,
                              hipStream_t stream)
{
  (void)in_sizes; (void)n_in; (void)out_size; (void)ws_size;

  // Leaf order: [src_tokens, prev_tgt_tokens, params-tree (sorted dict keys)]
  const int* src_tok = (const int*)d_in[0];
  const int* tgt_tok = (const int*)d_in[1];
  auto F = [&](int i) { return (const float*)d_in[i]; };
  const int EMB_SRC = 54, EMB_TGT = 55, ENC_LN_B = 84, ENC_LN_G = 85,
            OUT_B = 86, OUT_W = 87, PE = 88;

  // ---- workspace arena (fp32), decoder region reuses encoder region ----
  float* W  = (float*)d_ws;
  float* xe = W + 0;          // B*S*D
  float* xn = W + 1048576;    // B*S*D
  float* u  = W + 2097152;    // B*S*E
  float* vv = W + 5242880;    // B*S*E
  float* zb = W + 8388608;    // B*S*SZ
  float* qb = W + 8650752;    // B*S*SZ
  float* kb = W + 8912896;    // B*S*SZ
  float* qk = W + 9175040;    // B*S*S
  float* at = W + 10223616;   // B*S*E
  float* ua = W + 13369344;   // B*S*E
  // decoder phase (overlaps encoder scratch)
  float* y  = W + 0;          // B*T*D
  float* yn = W + 1048576;
  float* Qb = W + 2097152;
  float* Kb = W + 3145728;
  float* Vb = W + 4194304;
  float* sc = W + 5242880;    // B*H*T*T
  float* Ob = W + 13631488;   // B*T*D
  float* hb = W + 14680064;   // B*T*DFF
  // survives both phases
  float* enc_out = W + 18874368;                           // B*S*D
  unsigned short* wbuf = (unsigned short*)(W + 19922944);  // bf16 weight scratch

  const long nBSD = (long)B_ * S_ * D_;
  const long nBSE = (long)B_ * S_ * E_;
  const long nBSZ = (long)B_ * S_ * SZ_;
  const long nBSS = (long)B_ * S_ * S_;
  const float ESCALE = 22.62741699796952f;   // sqrt(512)

  auto ew = [](long n) { return dim3((unsigned)((n + 255) / 256)); };

  // activation-B GEMM (fp32 B staged with on-the-fly convert)
  auto gemm = [&](const float* A, const float* Bm, const float* bias, const float* Res,
                  float* C, int M, int N, int K, int lda, int ldb, int ldc,
                  long sAb, long sBb, long sCb, long sAh, long sBh, long sCh,
                  int Hn, int nz, int flags) {
    dim3 g(N / 64, M / 128, nz);
    launch_gemm(flags, g, stream, A, Bm, bias, Res, C, M, N, K, lda, ldb, ldc,
                sAb, sBb, sCb, sAh, sBh, sCh, Hn);
  };
  // weight GEMM: convert weight fp32[K,N] -> bf16[N,K] once, then async-staged GEMM
  auto wgemm = [&](const float* A, const float* Wf, const float* bias, const float* Res,
                   float* C, int M, int N, int K, int lda, int ldc, int flags) {
    dim3 gt(N / 32, K / 32, 1);
    convT_kernel<<<gt, 256, 0, stream>>>(Wf, wbuf, K, N);
    dim3 g(N / 64, M / 128, 1);
    launch_gemm(flags | GF_BF16B, g, stream, A, wbuf, bias, Res, C, M, N, K, lda, K, ldc,
                0, 0, 0, 0, 0, 0, 1);
  };

  // ================= encoder =================
  embed_kernel<<<ew(nBSD), 256, 0, stream>>>(src_tok, F(EMB_SRC), F(PE), xe, S_, D_, ESCALE, nBSD);
  for (int l = 0; l < 2; ++l) {
    int e = 56 + l * 14;  // bk,bq,gk,gq,ln.b,ln.g,wo.b,wo.w,wu.b,wu.w,wv.b,wv.w,wz.b,wz.w
    layernorm_kernel<<<B_ * S_, 256, 0, stream>>>(xe, F(e + 5), F(e + 4), xn, D_);
    wgemm(xn, F(e + 9),  F(e + 8),  nullptr, u,  2048, E_,  D_, D_, E_,  GF_BIAS);
    wgemm(xn, F(e + 11), F(e + 10), nullptr, vv, 2048, E_,  D_, D_, E_,  GF_BIAS);
    wgemm(xn, F(e + 13), F(e + 12), nullptr, zb, 2048, SZ_, D_, D_, SZ_, GF_BIAS);
    qkgate_kernel<<<ew(nBSZ), 256, 0, stream>>>(zb, F(e + 3), F(e + 1), F(e + 2), F(e + 0), qb, kb, nBSZ);
    // qk[b] = q[b] @ k[b]^T   (M=N=512, K=128, batched over B)
    gemm(qb, kb, nullptr, nullptr, qk, S_, S_, SZ_, SZ_, SZ_, S_,
         (long)S_ * SZ_, (long)S_ * SZ_, (long)S_ * S_, 0, 0, 0, 1, B_, GF_TRANSB);
    relusq_mask_kernel<<<ew(nBSS), 256, 0, stream>>>(qk, src_tok, nBSS);
    // attn[b] = a[b] @ v[b]   (M=512, N=1536, K=512)
    gemm(qk, vv, nullptr, nullptr, at, S_, E_, S_, S_, E_, E_,
         (long)S_ * S_, (long)S_ * E_, (long)S_ * E_, 0, 0, 0, 1, B_, 0);
    mul_kernel<<<ew(nBSE), 256, 0, stream>>>(u, at, ua, nBSE);
    wgemm(ua, F(e + 7), F(e + 6), xe, xe, 2048, D_, E_, E_, D_, GF_BIAS | GF_RES);
  }
  layernorm_kernel<<<B_ * S_, 256, 0, stream>>>(xe, F(ENC_LN_G), F(ENC_LN_B), enc_out, D_);

  // ================= decoder =================
  embed_kernel<<<ew(nBSD), 256, 0, stream>>>(tgt_tok, F(EMB_TGT), F(PE), y, T_, D_, ESCALE, nBSD);

  auto mha_block = [&](int mb, const float* kvsrc, const int* ktok, int causal) {
    // mb -> k.b,k.w,o.b,o.w,q.b,q.w,v.b,v.w
    wgemm(yn,    F(mb + 5), F(mb + 4), nullptr, Qb, 2048, D_, D_, D_, D_, GF_BIAS);
    wgemm(kvsrc, F(mb + 1), F(mb + 0), nullptr, Kb, 2048, D_, D_, D_, D_, GF_BIAS);
    wgemm(kvsrc, F(mb + 7), F(mb + 6), nullptr, Vb, 2048, D_, D_, D_, D_, GF_BIAS);
    // scores[b,h] = Q[b,:,h*64:] @ K[b,:,h*64:]^T   (M=N=512, K=64, 32 batches)
    gemm(Qb, Kb, nullptr, nullptr, sc, T_, T_, HD_, D_, D_, T_,
         (long)T_ * D_, (long)T_ * D_, (long)H_ * T_ * T_,
         HD_, HD_, (long)T_ * T_, H_, B_ * H_, GF_TRANSB);
    softmax_kernel<<<B_ * H_ * T_, 256, 0, stream>>>(sc, ktok, T_, T_, H_, causal, 0.125f);
    // O[b,:,h*64:] = attn[b,h] @ V[b,:,h*64:]   (M=512, N=64, K=512)
    gemm(sc, Vb, nullptr, nullptr, Ob, T_, HD_, T_, T_, D_, D_,
         (long)H_ * T_ * T_, (long)T_ * D_, (long)T_ * D_,
         (long)T_ * T_, HD_, HD_, H_, B_ * H_, 0);
    wgemm(Ob, F(mb + 3), F(mb + 2), y, y, 2048, D_, D_, D_, D_, GF_BIAS | GF_RES);
  };

  for (int l = 0; l < 2; ++l) {
    int d0 = 2 + l * 26;  // ca(8), ffn(4), ln1(2), ln2(2), ln3(2), sa(8)
    layernorm_kernel<<<B_ * T_, 256, 0, stream>>>(y, F(d0 + 13), F(d0 + 12), yn, D_);
    mha_block(d0 + 18, yn, tgt_tok, 1);       // self-attention (causal)
    layernorm_kernel<<<B_ * T_, 256, 0, stream>>>(y, F(d0 + 15), F(d0 + 14), yn, D_);
    mha_block(d0 + 0, enc_out, src_tok, 0);   // cross-attention
    layernorm_kernel<<<B_ * T_, 256, 0, stream>>>(y, F(d0 + 17), F(d0 + 16), yn, D_);
    wgemm(yn, F(d0 + 9),  F(d0 + 8),  nullptr, hb, 2048, DFF_, D_,  D_,  DFF_,
          GF_BIAS | GF_RELU);
    wgemm(hb, F(d0 + 11), F(d0 + 10), y, y,    2048, D_,  DFF_, DFF_, D_,
          GF_BIAS | GF_RES);
  }

  // ================= output projection (dominant GEMM) =================
  wgemm(y, F(OUT_W), F(OUT_B), nullptr, (float*)d_out, 2048, V_, D_, D_, V_, GF_BIAS);
}
